// AttentionBlock_31791347925475
// MI455X (gfx1250) — matrix operand order repaired and checked by reading, not compiled, not run on hardware
//
#include <hip/hip_runtime.h>
#include <hip/hip_bf16.h>

// ---------------------------------------------------------------------------
// Non-local attention block, flash-attention style, CDNA5 WMMA (gfx1250).
// B=8, C=64, C8=8, N=H*W=4096.
//
// Pass 1 (proj_kernel): f = Wf x + bf, g = Wg x + bg, h = Wh x + bh,
//                       written bf16 into workspace ([B][rows][N], row-major N).
// Pass 2 (attn_kernel): per wave: one 16-query i-tile. Loop over 64-key
//   j-tiles: st[j,i] = sum_c g[c,j] f[c,i] via 4x WMMA bf16 (K=32, c padded),
//   online softmax (stats per lane, lane = column i), then
//   o[c,i] += sum_j h[c,j] p[j,i] via 8x WMMA bf16. Epilogue: x + o/l.
//   Cross-half (xor-16) exchanges use v_permlanex16_b32 (VALU) instead of
//   ds_bpermute + s_wait_dscnt round-trips.
// ---------------------------------------------------------------------------

typedef __attribute__((ext_vector_type(16))) __bf16 v16bf;
typedef __attribute__((ext_vector_type(8)))  float  v8f;

#define NB   8
#define CH   64
#define C8   8
#define NN   4096   // H*W
#define JT   64     // keys per block iteration

// xor-16 half-wave exchange for wave32: lane k of each half reads lane k of
// the other half. Identity nibble selectors 0x76543210 / 0xfedcba98.
__device__ __forceinline__ float shflx16(float v) {
#if __has_builtin(__builtin_amdgcn_permlanex16)
  int i = __builtin_bit_cast(int, v);
  int r = __builtin_amdgcn_permlanex16(i, i, 0x76543210, 0xfedcba98, false, false);
  return __builtin_bit_cast(float, r);
#else
  return __shfl_xor(v, 16, 32);
#endif
}

// ========================= Pass 1: projections =============================
__global__ __launch_bounds__(256)
void proj_kernel(const float* __restrict__ x,
                 const float* __restrict__ Wf, const float* __restrict__ bf,
                 const float* __restrict__ Wg, const float* __restrict__ bg,
                 const float* __restrict__ Wh, const float* __restrict__ bh,
                 __bf16* __restrict__ fo, __bf16* __restrict__ go,
                 __bf16* __restrict__ ho) {
  __shared__ float wl[80 * 64];   // rows 0..7 Wf, 8..15 Wg, 16..79 Wh
  __shared__ float bl[80];
  const int tid = threadIdx.x;

  for (int i = tid; i < 512;  i += 256) wl[i]        = Wf[i];
  for (int i = tid; i < 512;  i += 256) wl[512 + i]  = Wg[i];
  for (int i = tid; i < 4096; i += 256) wl[1024 + i] = Wh[i];
  if (tid < 8)        bl[tid] = bf[tid];
  else if (tid < 16)  bl[tid] = bg[tid - 8];
  else if (tid < 80)  bl[tid] = bh[tid - 16];
  __syncthreads();

  const int gid = blockIdx.x * 256 + tid;     // 0 .. B*N-1
  const int b   = gid >> 12;
  const int n   = gid & (NN - 1);

  float xv[CH];
#pragma unroll
  for (int c = 0; c < CH; ++c)
    xv[c] = x[((size_t)(b * CH + c)) * NN + n];

#pragma unroll
  for (int o = 0; o < C8; ++o) {
    float af = bl[o], ag = bl[8 + o];
#pragma unroll
    for (int c = 0; c < CH; ++c) {
      af += wl[o * 64 + c]       * xv[c];
      ag += wl[512 + o * 64 + c] * xv[c];
    }
    fo[((size_t)(b * C8 + o)) * NN + n] = (__bf16)af;
    go[((size_t)(b * C8 + o)) * NN + n] = (__bf16)ag;
  }
  for (int o = 0; o < CH; ++o) {
    float ah = bl[16 + o];
#pragma unroll
    for (int c = 0; c < CH; ++c) ah += wl[1024 + o * 64 + c] * xv[c];
    ho[((size_t)(b * CH + o)) * NN + n] = (__bf16)ah;
  }
}

// ===================== Pass 2: fused flash attention =======================
// Block = 256 threads (8 wave32). Wave w handles i-tile = blockIdx.x*8 + w.
// 256 i-tiles per batch, 32 blocks per batch -> whole block shares batch b,
// so g/h j-tiles are staged once in LDS and reused by all 8 waves.
__global__ __launch_bounds__(256)
void attn_kernel(const float* __restrict__ x,
                 const __bf16* __restrict__ fw,
                 const __bf16* __restrict__ gw,
                 const __bf16* __restrict__ hw,
                 float* __restrict__ out) {
  __shared__ __bf16 g_lds[C8][JT];
  __shared__ __bf16 h_lds[CH][JT + 2];   // +2 pad to spread LDS banks

  const int tid  = threadIdx.x;
  const int lane = tid & 31;
  const int wave = tid >> 5;
  const int grp  = lane >> 4;        // lane half (0 or 1)
  const int li   = lane & 15;        // column index i within the tile
  const int tile = blockIdx.x * 8 + wave;
  const int b    = tile >> 8;
  const int i0   = (tile & 255) << 4;

  // B operand: f (K = c padded to 32, N = i). k order used here (contiguous):
  // element e of lane L holds B[K = e + 16*(L/16), N = L%16].
  v16bf fB;
#pragma unroll
  for (int e = 0; e < 16; ++e) {
    const int K = e + 16 * grp;
    fB[e] = (K < C8) ? fw[((size_t)(b * C8 + K)) * NN + i0 + li] : (__bf16)0.0f;
  }

  v8f acc[4];
#pragma unroll
  for (int cc = 0; cc < 4; ++cc) acc[cc] = (v8f){};
  float mrow = -1e30f, lrow = 0.0f;

  for (int j0 = 0; j0 < NN; j0 += JT) {
    __syncthreads();
    // ---- cooperative LDS staging of the 64-key j-tile ----
    {                                     // g: 8 rows x 64 keys, 2 per thread
      const int r = tid >> 5, s = (tid & 31) * 2;
      const __bf16* src = gw + ((size_t)(b * C8 + r)) * NN + j0 + s;
#pragma unroll
      for (int q = 0; q < 2; ++q) g_lds[r][s + q] = src[q];
    }
    {                                     // h: 64 rows x 64 keys, 16 per thread
      const int row = tid >> 2, seg = (tid & 3) * 16;
      const __bf16* src = hw + ((size_t)(b * CH + row)) * NN + j0 + seg;
#pragma unroll
      for (int q = 0; q < 16; ++q) h_lds[row][seg + q] = src[q];
    }
    if (j0 + JT < NN) {                   // gfx1250 global_prefetch_b8
      __builtin_prefetch(hw + ((size_t)(b * CH + (tid >> 2))) * NN + j0 + JT, 0, 1);
    }
    __syncthreads();

    // ---- scores: st[j,i] = sum_c g[c,j] f[c,i], four 16x16 tiles ----
    // Documented 16-bit A layout: elem e -> K = (e&7) + 8*grp + 16*(e>>3),
    // M = li. K < 8 (real c) only for grp==0 && e<8.
    v8f d[4];
#pragma unroll
    for (int jt = 0; jt < 4; ++jt) {
      v16bf gA;
#pragma unroll
      for (int e = 0; e < 16; ++e) {
        const bool valid = (grp == 0) && (e < 8);
        gA[e] = valid ? g_lds[e][jt * 16 + li] : (__bf16)0.0f;
      }
      v8f z = (v8f){};
      d[jt] = __builtin_amdgcn_wmma_f32_16x16x32_bf16(false, gA, false, fB,
                                                      (short)0, z, false, false);
    }

    // ---- online softmax: lane holds column i = li; rows j spread over
    // 8 regs x 4 tiles in-lane + the cross-half partner (permlanex16) ----
    float tmax = -1e30f;
#pragma unroll
    for (int jt = 0; jt < 4; ++jt)
#pragma unroll
      for (int r = 0; r < 8; ++r) tmax = fmaxf(tmax, d[jt][r]);
    tmax = fmaxf(tmax, shflx16(tmax));
    const float mnew  = fmaxf(mrow, tmax);
    const float alpha = __expf(mrow - mnew);
    mrow = mnew;

    float tsum = 0.0f;
#pragma unroll
    for (int jt = 0; jt < 4; ++jt)
#pragma unroll
      for (int r = 0; r < 8; ++r) {
        d[jt][r] = __expf(d[jt][r] - mnew);   // p in place
        tsum += d[jt][r];
      }
    tsum += shflx16(tsum);
    lrow = lrow * alpha + tsum;

#pragma unroll
    for (int cc = 0; cc < 4; ++cc)
#pragma unroll
      for (int r = 0; r < 8; ++r) acc[cc][r] *= alpha;

    // ---- pack P (64j x 16i) into two B operands via cross-half swap ----
    v16bf pB[2];
#pragma unroll
    for (int jh = 0; jh < 2; ++jh) {
#pragma unroll
      for (int r = 0; r < 8; ++r) {
        const float a  = d[2 * jh][r];        // local j' = r + 8*grp
        const float bb = d[2 * jh + 1][r];    // local j' = 16 + r + 8*grp
        const float xa = shflx16(a);
        const float xb = shflx16(bb);
        pB[jh][r]     = (__bf16)(grp ? xb : a);
        pB[jh][r + 8] = (__bf16)(grp ? bb : xa);
      }
    }

    // ---- o[c,i] += h[c,j] * P[j,i]: 4 channel chunks x 2 key halves ----
#pragma unroll
    for (int cc = 0; cc < 4; ++cc) {
#pragma unroll
      for (int jh = 0; jh < 2; ++jh) {
        v16bf hA;
#pragma unroll
        for (int e = 0; e < 16; ++e) {
          const int K = e + 16 * grp;   // j in 32-half
          hA[e] = h_lds[cc * 16 + li][jh * 32 + K];
        }
        acc[cc] = __builtin_amdgcn_wmma_f32_16x16x32_bf16(false, hA, false,
                                                          pB[jh], (short)0,
                                                          acc[cc], false, false);
      }
    }
  }

  // ---- epilogue: out = x + o / l (l is lane-uniform per column i) ----
  const float inv = 1.0f / lrow;
#pragma unroll
  for (int cc = 0; cc < 4; ++cc) {
#pragma unroll
    for (int r = 0; r < 8; ++r) {
      const int row = cc * 16 + r + 8 * grp;        // C/D layout: M = r+8*grp
      const size_t idx = ((size_t)(b * CH + row)) * NN + i0 + li;
      out[idx] = x[idx] + acc[cc][r] * inv;
    }
  }
}

// ============================== launcher ===================================
extern "C" void kernel_launch(void* const* d_in, const int* in_sizes, int n_in,
                              void* d_out, int out_size, void* d_ws, size_t ws_size,
                              hipStream_t stream) {
  const float* x  = (const float*)d_in[0];
  const float* Wf = (const float*)d_in[1];
  const float* bf = (const float*)d_in[2];
  const float* Wg = (const float*)d_in[3];
  const float* bg = (const float*)d_in[4];
  const float* Wh = (const float*)d_in[5];
  const float* bh = (const float*)d_in[6];
  float* out = (float*)d_out;

  // Workspace layout (bf16): f [B*8*N], g [B*8*N], h [B*64*N]  (~5 MB total)
  __bf16* f_ws = (__bf16*)d_ws;
  __bf16* g_ws = f_ws + (size_t)NB * C8 * NN;
  __bf16* h_ws = g_ws + (size_t)NB * C8 * NN;

  // Pass 1: B*N threads, 256/block -> 128 blocks
  proj_kernel<<<(NB * NN) / 256, 256, 0, stream>>>(x, Wf, bf, Wg, bg, Wh, bh,
                                                   f_ws, g_ws, h_ws);
  // Pass 2: B*N/16 i-tiles, 8 waves/block -> 256 blocks
  attn_kernel<<<(NB * NN / 16) / 8, 256, 0, stream>>>(x, f_ws, g_ws, h_ws, out);
}
